// RouteNet_72069551227087
// MI455X (gfx1250) — matrix operand order repaired and checked by reading, not compiled, not run on hardware
//
#include <hip/hip_runtime.h>
#include <cstdint>

// ---------------------------------------------------------------------------
// RouteNet on MI455X (gfx1250): bf16 WMMA GEMMs with f32 accumulate.
// ~3 TFLOP of path-GRU GEMM dominates (compute bound; weights L2-resident).
// Inner loop pipeline (per 32-wide K chunk):
//   - B weight tiles: global -> LDS via global_load_async_to_lds_b128,
//     TRIPLE-buffered, shared by the block's 4 waves, synchronized with
//     own-wave s_wait_asynccnt + a RAW split barrier (s_barrier_signal/wait
//     via inline asm, no fence -> no loadcnt drain at the barrier).
//   - A tiles (gathered x row / hidden row): raw f32 loads issued one chunk
//     ahead; with the fence-free barrier they genuinely overlap the WMMAs.
//   - 6 x v_wmma_f32_16x16x32_bf16 (r/z/n gates for W_ih and W_hh).
// GRU gate math + scatter-add into link aggregation fused in the epilogue.
// ---------------------------------------------------------------------------

typedef __bf16 bf16_t;
typedef __attribute__((ext_vector_type(16))) __bf16 v16bf;
typedef __attribute__((ext_vector_type(8)))  float  v8f;

#define WMMA_BF16(A, B, C) \
  __builtin_amdgcn_wmma_f32_16x16x32_bf16(false, (A), false, (B), (short)0, (C), false, false)

static __device__ __forceinline__ float sigmoidf_(float x) {
  return 1.0f / (1.0f + __expf(-x));
}
static __device__ __forceinline__ float tanhf_(float x) {
  float e = __expf(-2.0f * x);
  return (1.0f - e) / (1.0f + e);
}

// Raw workgroup split barrier: execution sync only (compiler-level memory
// clobber so LDS reads are not hoisted, but NO hardware fence -> the memory
// legalizer does not drain loadcnt here; producer ordering is provided by the
// explicit s_wait_asynccnt before the signal).
static __device__ __forceinline__ void wg_barrier() {
  asm volatile("s_barrier_signal -1\n\t"
               "s_barrier_wait -1" ::: "memory");
}

// Async 16B copy global -> LDS (ASYNCcnt-tracked).
static __device__ __forceinline__ void async_copy16(uint32_t lds_off, uint64_t gaddr) {
  asm volatile("global_load_async_to_lds_b128 %0, %1, off"
               :: "v"(lds_off), "v"(gaddr));
}

// Raw A-tile chunk (16 f32 per lane) per ISA 7.12.2 16-bit A layout:
// lane reads 8 floats at k0+koff and 8 floats at k0+16+koff, koff=(lane>>4)*8.
struct RawA { float4 q0, q1, q2, q3; };

static __device__ __forceinline__ RawA load_rawA(const float* __restrict__ row,
                                                 int k0, int koff) {
  RawA r;
  const float4* p0 = (const float4*)(row + k0 + koff);
  const float4* p1 = (const float4*)(row + k0 + 16 + koff);
  r.q0 = p0[0]; r.q1 = p0[1]; r.q2 = p1[0]; r.q3 = p1[1];
  return r;
}

static __device__ __forceinline__ v16bf cvtA(const RawA& r) {
  v16bf a;
  a[0]  = (bf16_t)r.q0.x; a[1]  = (bf16_t)r.q0.y; a[2]  = (bf16_t)r.q0.z; a[3]  = (bf16_t)r.q0.w;
  a[4]  = (bf16_t)r.q1.x; a[5]  = (bf16_t)r.q1.y; a[6]  = (bf16_t)r.q1.z; a[7]  = (bf16_t)r.q1.w;
  a[8]  = (bf16_t)r.q2.x; a[9]  = (bf16_t)r.q2.y; a[10] = (bf16_t)r.q2.z; a[11] = (bf16_t)r.q2.w;
  a[12] = (bf16_t)r.q3.x; a[13] = (bf16_t)r.q3.y; a[14] = (bf16_t)r.q3.z; a[15] = (bf16_t)r.q3.w;
  return a;
}

// ---------------------------------------------------------------------------
// Weight pre-tiling: W (torch layout [N,K] f32) -> bf16 tiles so a B tile
// (32x16, K x N) is one contiguous 32B chunk per lane:
//   flat = ((kt*ntiles + nt)*32 + lane)*16 + e
//   n = nt*16 + (lane&15), k = kt*32 + (lane>>4)*8 + (e&7) + (e>>3)*16    (B[k][n] = W[n][k])
// ---------------------------------------------------------------------------
__global__ void pretile_kernel(const float* __restrict__ W, bf16_t* __restrict__ Wt,
                               int K, int N) {
  int total = K * N;
  int ntiles = N >> 4;
  for (int idx = blockIdx.x * blockDim.x + threadIdx.x; idx < total;
       idx += gridDim.x * blockDim.x) {
    int e    = idx & 15;
    int lane = (idx >> 4) & 31;
    int rest = idx >> 9;
    int nt   = rest % ntiles;
    int kt   = rest / ntiles;
    int n = nt * 16 + (lane & 15);
    int k = kt * 32 + (lane >> 4) * 8 + (e & 7) + (e >> 3) * 16;
    Wt[idx] = (bf16_t)W[(size_t)n * K + k];
  }
}

__global__ void init_state_kernel(float* __restrict__ h, const float* __restrict__ feat,
                                  int M) {
  int idx = blockIdx.x * blockDim.x + threadIdx.x;
  if (idx < M * 512) {
    int m = idx >> 9;
    int d = idx & 511;
    h[idx] = (d == 0) ? feat[m] : 0.0f;
  }
}

__global__ void zero_kernel(float* __restrict__ p, int n) {
  int idx = blockIdx.x * blockDim.x + threadIdx.x;
  if (idx < n) p[idx] = 0.0f;
}

// ---------------------------------------------------------------------------
// Fused GRU step (D=512, gates=1536). Block = 128 threads = 4 waves.
//   blockIdx.x in [0,32): 16-col tile of the new hidden state
//   blockIdx.y*64 + wave*16: 16-row tile
// ---------------------------------------------------------------------------
__global__ void __launch_bounds__(128) gru_step_kernel(
    const float*  __restrict__ xsrc,    // x source rows (link_h or agg)
    const int*    __restrict__ links,   // gather/scatter index or nullptr
    const float*  __restrict__ h_cur,
    float*        __restrict__ h_nxt,
    float*        __restrict__ agg,     // scatter target or nullptr
    const bf16_t* __restrict__ Wih_t,   // tiled bf16 [16][96][32][16]
    const bf16_t* __restrict__ Whh_t,
    const float*  __restrict__ bih,     // [1536]
    const float*  __restrict__ bhh,     // [1536]
    int s, int L, int M) {
  // 3 buffers x 6 tiles x 1KB (triple buffer -> one barrier per K chunk)
  __shared__ char smem[3 * 6144];

  const int tid   = threadIdx.x;
  const int lane  = tid & 31;
  const int wave  = tid >> 5;
  const int nt0   = blockIdx.x;
  const int n0    = nt0 * 16;
  const int mBase = blockIdx.y * 64 + wave * 16;

  const int mr = mBase + (lane & 15);
  const int mc = (mr < M) ? mr : (M - 1);     // clamp loads; predicate stores
  const float* xrow = links ? (xsrc + (size_t)links[mc * L + s] * 512)
                            : (xsrc + (size_t)mc * 512);
  const float* hrow = h_cur + (size_t)mc * 512;
  const int koff = (lane >> 4) * 8;

  // Per-thread async-stage chunks: 3 x 16B, flat byte = i*2048 + tid*16.
  // Tile g = flat>>10: g<3 -> Wih gate tile at nt = g*32+nt0, g>=3 -> Whh.
  const uint32_t ldsBase = (uint32_t)(uintptr_t)smem;
  uint32_t chunkFlat[3];
  uint64_t srcChunk[3];
#pragma unroll
  for (int i = 0; i < 3; ++i) {
    int flat = i * 2048 + tid * 16;
    int g = flat >> 10;
    int b = flat & 1023;
    const bf16_t* wb = (g < 3) ? Wih_t : Whh_t;
    int gg = (g < 3) ? g : g - 3;
    chunkFlat[i] = (uint32_t)flat;
    srcChunk[i]  = (uint64_t)(uintptr_t)((const char*)wb +
                    (size_t)(gg * 32 + nt0) * 1024 + b);
  }

  auto issue_stage = [&](int kt, int buf) {
    const uint64_t kOff = (uint64_t)kt * (96 * 1024);
    const uint32_t lOff = ldsBase + (uint32_t)buf * 6144;
#pragma unroll
    for (int i = 0; i < 3; ++i)
      async_copy16(lOff + chunkFlat[i], srcChunk[i] + kOff);
  };

  v8f acc_ir = {}, acc_iz = {}, acc_in = {};
  v8f acc_hr = {}, acc_hz = {}, acc_hn = {};

  RawA xr = load_rawA(xrow, 0, koff);
  RawA hr = load_rawA(hrow, 0, koff);
  issue_stage(0, 0);

  for (int kt = 0; kt < 16; ++kt) {
    RawA xr2, hr2;
    if (kt < 15) {
      xr2 = load_rawA(xrow, (kt + 1) * 32, koff);   // A for next chunk (no wait)
      hr2 = load_rawA(hrow, (kt + 1) * 32, koff);
      issue_stage(kt + 1, (kt + 1) % 3);            // B for next chunk -> LDS
      asm volatile("s_wait_asynccnt 0x3");          // own stage kt landed
    } else {
      asm volatile("s_wait_asynccnt 0x0");
    }
    wg_barrier();  // raw split barrier: stage kt visible, no loadcnt drain

    v16bf ax = cvtA(xr);
    v16bf ah = cvtA(hr);

    const char* bb = smem + (kt % 3) * 6144 + lane * 32;
    v16bf b0 = *(const v16bf*)(bb + 0 * 1024);
    v16bf b1 = *(const v16bf*)(bb + 1 * 1024);
    v16bf b2 = *(const v16bf*)(bb + 2 * 1024);
    v16bf b3 = *(const v16bf*)(bb + 3 * 1024);
    v16bf b4 = *(const v16bf*)(bb + 4 * 1024);
    v16bf b5 = *(const v16bf*)(bb + 5 * 1024);

    acc_ir = WMMA_BF16(ax, b0, acc_ir);
    acc_iz = WMMA_BF16(ax, b1, acc_iz);
    acc_in = WMMA_BF16(ax, b2, acc_in);
    acc_hr = WMMA_BF16(ah, b3, acc_hr);
    acc_hz = WMMA_BF16(ah, b4, acc_hz);
    acc_hn = WMMA_BF16(ah, b5, acc_hn);

    if (kt < 15) { xr = xr2; hr = hr2; }
  }

  // C/D layout: lanes 0-15 -> M=v, N=lane; lanes 16-31 -> M=8+v, N=lane-16
  const int gn = n0 + (lane & 15);
  const float b_ir = bih[gn], b_iz = bih[512 + gn], b_in = bih[1024 + gn];
  const float b_hr = bhh[gn], b_hz = bhh[512 + gn], b_hn = bhh[1024 + gn];
  const int mrow0 = mBase + (lane >> 4) * 8;

#pragma unroll
  for (int v = 0; v < 8; ++v) {
    const int gm  = mrow0 + v;
    const int gmc = (gm < M) ? gm : (M - 1);
    const float hold = h_cur[(size_t)gmc * 512 + gn];
    const float r  = sigmoidf_(acc_ir[v] + b_ir + acc_hr[v] + b_hr);
    const float z  = sigmoidf_(acc_iz[v] + b_iz + acc_hz[v] + b_hz);
    const float ng = tanhf_(acc_in[v] + b_in + r * (acc_hn[v] + b_hn));
    const float hn2 = (1.0f - z) * ng + z * hold;
    if (gm < M) {
      h_nxt[(size_t)gm * 512 + gn] = hn2;
      if (agg) {
        const int lk = links[gm * L + s];
        atomicAdd(agg + (size_t)lk * 512 + gn, hn2);
      }
    }
  }
}

// ---------------------------------------------------------------------------
// WMMA linear layer: out[M,N] = act( in[M,K] @ Wt + bias ), bf16 x bf16 -> f32
// ---------------------------------------------------------------------------
__global__ void __launch_bounds__(128) linear_kernel(
    const float*  __restrict__ in,    // [M,K] f32
    const bf16_t* __restrict__ Wt,    // tiled bf16 [K/32][N/16][32][16]
    const float*  __restrict__ bias,  // [N]
    float*        __restrict__ out,   // [M,N]
    int M, int K, int N, int do_relu) {
  const int lane  = threadIdx.x & 31;
  const int wave  = threadIdx.x >> 5;
  const int nt    = blockIdx.x;
  const int n0    = nt * 16;
  const int mBase = blockIdx.y * 64 + wave * 16;
  const int ntiles = N >> 4;
  const int ktiles = K >> 5;

  const int mr = mBase + (lane & 15);
  const int mc = (mr < M) ? mr : (M - 1);
  const float* arow = in + (size_t)mc * K;
  const int koff = (lane >> 4) * 8;

  v8f acc = {};
#pragma unroll 4
  for (int kt = 0; kt < ktiles; ++kt) {
    RawA ar = load_rawA(arow, kt * 32, koff);
    v16bf b = *(const v16bf*)(Wt + (((size_t)(kt * ntiles + nt) << 5) + lane) * 16);
    v16bf a = cvtA(ar);
    acc = WMMA_BF16(a, b, acc);
  }

  const int gn = n0 + (lane & 15);
  const float bs = bias[gn];
  const int mrow0 = mBase + (lane >> 4) * 8;
#pragma unroll
  for (int v = 0; v < 8; ++v) {
    const int gm = mrow0 + v;
    if (gm < M) {
      float r = acc[v] + bs;
      if (do_relu) r = fmaxf(r, 0.0f);
      out[(size_t)gm * N + gn] = r;
    }
  }
}

__global__ void dot_out_kernel(const float* __restrict__ in, const float* __restrict__ w,
                               const float* __restrict__ b, float* __restrict__ out,
                               int M, int K) {
  int m = blockIdx.x * blockDim.x + threadIdx.x;
  if (m >= M) return;
  float s = 0.0f;
#pragma unroll 8
  for (int k = 0; k < K; ++k) s += in[(size_t)m * K + k] * w[k];
  out[m] = s + b[0];
}

// ---------------------------------------------------------------------------
// Host orchestration
// ---------------------------------------------------------------------------
extern "C" void kernel_launch(void* const* d_in, const int* in_sizes, int n_in,
                              void* d_out, int out_size, void* d_ws, size_t ws_size,
                              hipStream_t stream) {
  const int*   links         = (const int*)d_in[0];
  const float* link_capacity = (const float*)d_in[3];
  const float* bandwidth     = (const float*)d_in[4];
  const float* Wih_p = (const float*)d_in[5];
  const float* Whh_p = (const float*)d_in[6];
  const float* bih_p = (const float*)d_in[7];
  const float* bhh_p = (const float*)d_in[8];
  const float* Wih_l = (const float*)d_in[9];
  const float* Whh_l = (const float*)d_in[10];
  const float* bih_l = (const float*)d_in[11];
  const float* bhh_l = (const float*)d_in[12];
  const float* d_w1 = (const float*)d_in[13]; const float* d_b1 = (const float*)d_in[14];
  const float* d_w2 = (const float*)d_in[15]; const float* d_b2 = (const float*)d_in[16];
  const float* d_w3 = (const float*)d_in[17]; const float* d_b3 = (const float*)d_in[18];
  const float* j_w1 = (const float*)d_in[19]; const float* j_b1 = (const float*)d_in[20];
  const float* j_w2 = (const float*)d_in[21]; const float* j_b2 = (const float*)d_in[22];
  const float* j_w3 = (const float*)d_in[23]; const float* j_b3 = (const float*)d_in[24];

  const int n_links = in_sizes[3];
  const int n_paths = in_sizes[4];
  const int L       = in_sizes[0] / n_paths;   // 6
  const int D  = 512, G3 = 3 * 512, R1 = 256, R2 = 128, T = 8;

  // workspace carve-up
  char* ws = (char*)d_ws;
  size_t off = 0;
  auto carve = [&](size_t bytes) -> char* {
    char* p = ws + off;
    off = (off + bytes + 255) & ~(size_t)255;
    return p;
  };
  float*  pathA  = (float*)carve((size_t)n_paths * D * 4);
  float*  pathB  = (float*)carve((size_t)n_paths * D * 4);
  float*  linkA  = (float*)carve((size_t)n_links * D * 4);
  float*  linkB  = (float*)carve((size_t)n_links * D * 4);
  float*  aggB   = (float*)carve((size_t)n_links * D * 4);
  bf16_t* Wp_ih  = (bf16_t*)carve((size_t)G3 * D * 2);
  bf16_t* Wp_hh  = (bf16_t*)carve((size_t)G3 * D * 2);
  bf16_t* Wl_ih  = (bf16_t*)carve((size_t)G3 * D * 2);
  bf16_t* Wl_hh  = (bf16_t*)carve((size_t)G3 * D * 2);
  bf16_t* dW1t   = (bf16_t*)carve((size_t)D * R1 * 2);
  bf16_t* dW2t   = (bf16_t*)carve((size_t)R1 * R2 * 2);
  bf16_t* jW1t   = (bf16_t*)carve((size_t)D * R1 * 2);
  bf16_t* jW2t   = (bf16_t*)carve((size_t)R1 * R2 * 2);
  float*  act1   = (float*)carve((size_t)n_paths * R1 * 4);
  float*  act2   = (float*)carve((size_t)n_paths * R2 * 4);

  // weight conversion / pre-tiling (bf16, WMMA B-operand lane layout)
  {
    int tot = G3 * D, blk = 256, g = (tot + blk - 1) / blk;
    pretile_kernel<<<g, blk, 0, stream>>>(Wih_p, Wp_ih, D, G3);
    pretile_kernel<<<g, blk, 0, stream>>>(Whh_p, Wp_hh, D, G3);
    pretile_kernel<<<g, blk, 0, stream>>>(Wih_l, Wl_ih, D, G3);
    pretile_kernel<<<g, blk, 0, stream>>>(Whh_l, Wl_hh, D, G3);
    int t1 = D * R1, g1 = (t1 + blk - 1) / blk;
    pretile_kernel<<<g1, blk, 0, stream>>>(d_w1, dW1t, D, R1);
    pretile_kernel<<<g1, blk, 0, stream>>>(j_w1, jW1t, D, R1);
    int t2 = R1 * R2, g2 = (t2 + blk - 1) / blk;
    pretile_kernel<<<g2, blk, 0, stream>>>(d_w2, dW2t, R1, R2);
    pretile_kernel<<<g2, blk, 0, stream>>>(j_w2, jW2t, R1, R2);
  }

  // state init: h[:,0] = feature, rest 0; agg = 0
  init_state_kernel<<<(n_paths * D + 255) / 256, 256, 0, stream>>>(pathA, bandwidth, n_paths);
  init_state_kernel<<<(n_links * D + 255) / 256, 256, 0, stream>>>(linkA, link_capacity, n_links);
  zero_kernel<<<(n_links * D + 255) / 256, 256, 0, stream>>>(aggB, n_links * D);

  float* pc = pathA; float* pn = pathB;
  float* lc = linkA; float* ln = linkB;

  for (int it = 0; it < T; ++it) {
    // path GRU scan over the L gathered link states; scatter-add messages
    for (int s = 0; s < L; ++s) {
      dim3 grid(D / 16, (n_paths + 63) / 64);
      gru_step_kernel<<<grid, 128, 0, stream>>>(lc, links, pc, pn, aggB,
                                                Wp_ih, Wp_hh, bih_p, bhh_p,
                                                s, L, n_paths);
      float* t = pc; pc = pn; pn = t;
    }
    // link GRU: link_h = gru(agg, link_h)
    {
      dim3 grid(D / 16, (n_links + 63) / 64);
      gru_step_kernel<<<grid, 128, 0, stream>>>(aggB, nullptr, lc, ln, nullptr,
                                                Wl_ih, Wl_hh, bih_l, bhh_l,
                                                0, L, n_links);
      float* t = lc; lc = ln; ln = t;
    }
    zero_kernel<<<(n_links * D + 255) / 256, 256, 0, stream>>>(aggB, n_links * D);
  }

  // readout MLPs on final path states
  float* out_d = (float*)d_out;
  float* out_j = out_d + n_paths;
  {
    dim3 g1(R1 / 16, (n_paths + 63) / 64);
    dim3 g2(R2 / 16, (n_paths + 63) / 64);
    // delay head
    linear_kernel<<<g1, 128, 0, stream>>>(pc, dW1t, d_b1, act1, n_paths, D, R1, 1);
    linear_kernel<<<g2, 128, 0, stream>>>(act1, dW2t, d_b2, act2, n_paths, R1, R2, 1);
    dot_out_kernel<<<(n_paths + 255) / 256, 256, 0, stream>>>(act2, d_w3, d_b3, out_d,
                                                              n_paths, R2);
    // jitter head
    linear_kernel<<<g1, 128, 0, stream>>>(pc, jW1t, j_b1, act1, n_paths, D, R1, 1);
    linear_kernel<<<g2, 128, 0, stream>>>(act1, jW2t, j_b2, act2, n_paths, R1, R2, 1);
    dot_out_kernel<<<(n_paths + 255) / 256, 256, 0, stream>>>(act2, j_w3, j_b3, out_j,
                                                              n_paths, R2);
  }
}
